// RNN_59682865545712
// MI455X (gfx1250) — matrix operand order repaired and checked by reading, not compiled, not run on hardware
//
#include <hip/hip_runtime.h>
#include <stdint.h>

// ---------------- problem constants ----------------
constexpr int B_  = 128;
constexpr int T_  = 512;
constexpr int DIN = 512;
constexpr int H_  = 2048;
constexpr int O_  = 1024;
constexpr int KA  = DIN + H_;   // 2560
constexpr int KB  = H_ + O_;    // 3072

// ---------------- WMMA types ----------------
typedef __attribute__((ext_vector_type(16))) __bf16 v16bf;
typedef __attribute__((ext_vector_type(8)))  float  v8f;

union Frag {
  v16bf v;
  uint4 q[2];
};
static_assert(sizeof(Frag) == 32, "frag size");

__device__ __forceinline__ uint16_t f2bf(float f) {
  // round-to-nearest-even fp32 -> bf16
  uint32_t u = __float_as_uint(f);
  uint32_t r = u + 0x7FFFu + ((u >> 16) & 1u);
  return (uint16_t)(r >> 16);
}

// ---------------- prep: fp32 -> bf16 (optionally scaled) ----------------
__global__ void convert_bf16(const float* __restrict__ src,
                             uint16_t* __restrict__ dst,
                             long n, float scale) {
  long i = (long)blockIdx.x * blockDim.x + threadIdx.x;
  long stride = (long)gridDim.x * blockDim.x;
  for (; i < n; i += stride) dst[i] = f2bf(src[i] * scale);
}

__global__ void zero_bar(unsigned* __restrict__ p) {
  p[threadIdx.x] = 0u;
}

// ---------------- device-wide barrier ----------------
__device__ __forceinline__ void grid_barrier(unsigned* cnt, unsigned* gen,
                                             unsigned nblocks) {
  __threadfence();          // release this block's global stores
  __syncthreads();
  if (threadIdx.x == 0) {
    unsigned g = __hip_atomic_load(gen, __ATOMIC_RELAXED, __HIP_MEMORY_SCOPE_AGENT);
    unsigned a = __hip_atomic_fetch_add(cnt, 1u, __ATOMIC_ACQ_REL, __HIP_MEMORY_SCOPE_AGENT);
    if (a == nblocks - 1u) {
      __hip_atomic_store(cnt, 0u, __ATOMIC_RELAXED, __HIP_MEMORY_SCOPE_AGENT);
      __hip_atomic_fetch_add(gen, 1u, __ATOMIC_RELEASE, __HIP_MEMORY_SCOPE_AGENT);
    } else {
      while (__hip_atomic_load(gen, __ATOMIC_ACQUIRE, __HIP_MEMORY_SCOPE_AGENT) == g)
        __builtin_amdgcn_s_sleep(1);
    }
  }
  __syncthreads();
}

// ---------------- persistent recurrent kernel ----------------
// A fragment (16x32 bf16): lane l -> row (l&15); two contiguous 8-elem (16B)
// runs at K = kc + (l>>4)*8 and K = kc + 16 + (l>>4)*8.
// B fragment (32x16 bf16): lane l -> column (l&15); two contiguous 8-elem (16B)
// runs at K = kc + (l>>4)*8 and K = kc + 16 + (l>>4)*8.  W stored (out,in) row-major => K-contig.
// C/D (16x16 f32): vgpr r, lane l -> element (M = (l>>4)*8 + r, N = l&15).
__global__ __launch_bounds__(256, 1)
void rnn_persist(const uint16_t* __restrict__ xbf,
                 const uint16_t* __restrict__ w1,
                 const uint16_t* __restrict__ w2,
                 const float* __restrict__ b1,
                 const float* __restrict__ b2,
                 uint16_t* __restrict__ hb0, uint16_t* __restrict__ hb1,
                 uint16_t* __restrict__ ob0, uint16_t* __restrict__ ob1,
                 float* __restrict__ out,
                 unsigned* __restrict__ bar) {
  const int lane   = threadIdx.x & 31;
  const int ln     = lane & 15;
  const int s      = lane >> 4;
  const int wid    = blockIdx.x * (blockDim.x >> 5) + (threadIdx.x >> 5);
  const int nwaves = gridDim.x * (blockDim.x >> 5);
  unsigned* cnt = bar;
  unsigned* gen = bar + 64;                 // separate cache line
  const long BTO = (long)B_ * T_ * O_;

  for (int t = 0; t < T_; ++t) {
    const uint16_t* hread = (t & 1) ? hb1 : hb0;
    uint16_t*       hwr   = (t & 1) ? hb0 : hb1;
    const uint16_t* oread = (t & 1) ? ob1 : ob0;
    uint16_t*       owr   = (t & 1) ? ob0 : ob1;

    // ---------- phase 1: h = relu([x_t, h] @ W1^T + b1) ----------
    constexpr int NT1 = (B_ / 16) * (H_ / 32);   // 512 wave-tiles
    for (int tile = wid; tile < NT1; tile += nwaves) {
      const int mt = tile & 7, nt = tile >> 3;
      const int m0 = mt * 16, n0 = nt * 32;
      const int row = m0 + ln;
      const uint16_t* xr  = xbf + ((long)row * T_ + t) * DIN;
      const uint16_t* hr  = hread + (long)row * H_;
      const uint16_t* wc0 = w1 + (long)(n0 + ln) * KA;
      const uint16_t* wc1 = w1 + (long)(n0 + 16 + ln) * KA;
      v8f acc0 = {}, acc1 = {};
      #pragma unroll 4
      for (int kc = 0; kc < DIN; kc += 32) {          // x part of K
        Frag a, p, q;
        a.q[0] = *(const uint4*)(xr + kc + s * 8);
        a.q[1] = *(const uint4*)(xr + kc + 16 + s * 8);
        p.q[0] = *(const uint4*)(wc0 + kc + s * 8);
        p.q[1] = *(const uint4*)(wc0 + kc + 16 + s * 8);
        q.q[0] = *(const uint4*)(wc1 + kc + s * 8);
        q.q[1] = *(const uint4*)(wc1 + kc + 16 + s * 8);
        acc0 = __builtin_amdgcn_wmma_f32_16x16x32_bf16(false, a.v, false, p.v,
                                                       (short)0, acc0, false, false);
        acc1 = __builtin_amdgcn_wmma_f32_16x16x32_bf16(false, a.v, false, q.v,
                                                       (short)0, acc1, false, false);
      }
      #pragma unroll 4
      for (int kc = 0; kc < H_; kc += 32) {           // h part of K
        Frag a, p, q;
        a.q[0] = *(const uint4*)(hr + kc + s * 8);
        a.q[1] = *(const uint4*)(hr + kc + 16 + s * 8);
        p.q[0] = *(const uint4*)(wc0 + DIN + kc + s * 8);
        p.q[1] = *(const uint4*)(wc0 + DIN + kc + 16 + s * 8);
        q.q[0] = *(const uint4*)(wc1 + DIN + kc + s * 8);
        q.q[1] = *(const uint4*)(wc1 + DIN + kc + 16 + s * 8);
        acc0 = __builtin_amdgcn_wmma_f32_16x16x32_bf16(false, a.v, false, p.v,
                                                       (short)0, acc0, false, false);
        acc1 = __builtin_amdgcn_wmma_f32_16x16x32_bf16(false, a.v, false, q.v,
                                                       (short)0, acc1, false, false);
      }
      const float bias0 = b1[n0 + ln], bias1 = b1[n0 + 16 + ln];
      #pragma unroll
      for (int r = 0; r < 8; ++r) {
        const int rr = m0 + s * 8 + r;
        float v0 = acc0[r] + bias0; v0 = v0 > 0.f ? v0 : 0.f;
        float v1 = acc1[r] + bias1; v1 = v1 > 0.f ? v1 : 0.f;
        hwr[(long)rr * H_ + n0 + ln]      = f2bf(v0);
        hwr[(long)rr * H_ + n0 + 16 + ln] = f2bf(v1);
      }
    }
    grid_barrier(cnt, gen, gridDim.x);

    // prefetch next timestep's x slab into cache while phase 2 runs
    if (t + 1 < T_) {
      const int gtid = blockIdx.x * blockDim.x + threadIdx.x;
      if (gtid < B_ * 8) {                    // 128 rows x 8 x 128B lines
        const int r = gtid >> 3, l = gtid & 7;
        __builtin_prefetch((const void*)(xbf + ((long)r * T_ + (t + 1)) * DIN + l * 64), 0, 1);
      }
    }

    // ---------- phase 2: o = relu([h, o] @ W2^T + b2) ----------
    constexpr int NT2 = (B_ / 16) * (O_ / 32);   // 256 wave-tiles
    for (int tile = wid; tile < NT2; tile += nwaves) {
      const int mt = tile & 7, nt = tile >> 3;
      const int m0 = mt * 16, n0 = nt * 32;
      const int row = m0 + ln;
      const uint16_t* hr  = hwr + (long)row * H_;      // h just produced
      const uint16_t* orr = oread + (long)row * O_;
      const uint16_t* wc0 = w2 + (long)(n0 + ln) * KB;
      const uint16_t* wc1 = w2 + (long)(n0 + 16 + ln) * KB;
      v8f acc0 = {}, acc1 = {};
      #pragma unroll 4
      for (int kc = 0; kc < H_; kc += 32) {           // h part of K
        Frag a, p, q;
        a.q[0] = *(const uint4*)(hr + kc + s * 8);
        a.q[1] = *(const uint4*)(hr + kc + 16 + s * 8);
        p.q[0] = *(const uint4*)(wc0 + kc + s * 8);
        p.q[1] = *(const uint4*)(wc0 + kc + 16 + s * 8);
        q.q[0] = *(const uint4*)(wc1 + kc + s * 8);
        q.q[1] = *(const uint4*)(wc1 + kc + 16 + s * 8);
        acc0 = __builtin_amdgcn_wmma_f32_16x16x32_bf16(false, a.v, false, p.v,
                                                       (short)0, acc0, false, false);
        acc1 = __builtin_amdgcn_wmma_f32_16x16x32_bf16(false, a.v, false, q.v,
                                                       (short)0, acc1, false, false);
      }
      #pragma unroll 4
      for (int kc = 0; kc < O_; kc += 32) {           // o part of K
        Frag a, p, q;
        a.q[0] = *(const uint4*)(orr + kc + s * 8);
        a.q[1] = *(const uint4*)(orr + kc + 16 + s * 8);
        p.q[0] = *(const uint4*)(wc0 + H_ + kc + s * 8);
        p.q[1] = *(const uint4*)(wc0 + H_ + kc + 16 + s * 8);
        q.q[0] = *(const uint4*)(wc1 + H_ + kc + s * 8);
        q.q[1] = *(const uint4*)(wc1 + H_ + kc + 16 + s * 8);
        acc0 = __builtin_amdgcn_wmma_f32_16x16x32_bf16(false, a.v, false, p.v,
                                                       (short)0, acc0, false, false);
        acc1 = __builtin_amdgcn_wmma_f32_16x16x32_bf16(false, a.v, false, q.v,
                                                       (short)0, acc1, false, false);
      }
      const float bias0 = b2[n0 + ln], bias1 = b2[n0 + 16 + ln];
      #pragma unroll
      for (int r = 0; r < 8; ++r) {
        const int rr = m0 + s * 8 + r;
        float v0 = acc0[r] + bias0; v0 = v0 > 0.f ? v0 : 0.f;
        float v1 = acc1[r] + bias1; v1 = v1 > 0.f ? v1 : 0.f;
        owr[(long)rr * O_ + n0 + ln]      = f2bf(v0);
        owr[(long)rr * O_ + n0 + 16 + ln] = f2bf(v1);
        const long oi0 = ((long)rr * T_ + t) * O_ + n0 + ln;
        const long oi1 = oi0 + 16;
        out[oi0] = v0;          out[oi1] = v1;          // first returned tensor
        out[BTO + oi0] = v0;    out[BTO + oi1] = v1;    // second (== first)
      }
    }
    grid_barrier(cnt, gen, gridDim.x);
  }
}

// ---------------- launch ----------------
extern "C" void kernel_launch(void* const* d_in, const int* in_sizes, int n_in,
                              void* d_out, int out_size, void* d_ws, size_t ws_size,
                              hipStream_t stream) {
  (void)in_sizes; (void)n_in; (void)out_size; (void)ws_size;
  const float* x  = (const float*)d_in[0];
  const float* h0 = (const float*)d_in[1];
  const float* o0 = (const float*)d_in[2];
  const float* W1 = (const float*)d_in[3];
  const float* b1 = (const float*)d_in[4];
  const float* W2 = (const float*)d_in[5];
  const float* b2 = (const float*)d_in[6];
  float* out = (float*)d_out;

  char* ws = (char*)d_ws;
  size_t off = 0;
  auto alloc = [&](size_t bytes) -> char* {
    off = (off + 255) & ~(size_t)255;
    char* p = ws + off;
    off += bytes;
    return p;
  };
  uint16_t* xbf  = (uint16_t*)alloc((size_t)B_ * T_ * DIN * 2);  // 67 MB
  uint16_t* w1bf = (uint16_t*)alloc((size_t)H_ * KA * 2);        // 10.5 MB
  uint16_t* w2bf = (uint16_t*)alloc((size_t)O_ * KB * 2);        // 6.3 MB
  uint16_t* hb0  = (uint16_t*)alloc((size_t)B_ * H_ * 2);
  uint16_t* hb1  = (uint16_t*)alloc((size_t)B_ * H_ * 2);
  uint16_t* ob0  = (uint16_t*)alloc((size_t)B_ * O_ * 2);
  uint16_t* ob1  = (uint16_t*)alloc((size_t)B_ * O_ * 2);
  unsigned* bar  = (unsigned*)alloc(512);

  const int thr = 256;
  convert_bf16<<<2048, thr, 0, stream>>>(x,  xbf,  (long)B_ * T_ * DIN, 1.0f);
  convert_bf16<<<1024, thr, 0, stream>>>(W1, w1bf, (long)H_ * KA, 1.0f);
  convert_bf16<<<1024, thr, 0, stream>>>(W2, w2bf, (long)O_ * KB, 1.0f);
  convert_bf16<<<256,  thr, 0, stream>>>(h0, hb0,  (long)B_ * H_, 45.254833996f); // sqrt(2048)
  convert_bf16<<<256,  thr, 0, stream>>>(o0, ob0,  (long)B_ * O_, 32.0f);          // sqrt(1024)
  zero_bar<<<1, 128, 0, stream>>>(bar);

  rnn_persist<<<128, 256, 0, stream>>>(xbf, w1bf, w2bf, b1, b2,
                                       hb0, hb1, ob0, ob1, out, bar);
}
